// HierarchicalGNNWithAttention_10685878632726
// MI455X (gfx1250) — matrix operand order repaired and checked by reading, not compiled, not run on hardware
//
#include <hip/hip_runtime.h>

// ---------------------------------------------------------------------------
// HierarchicalGNNWithAttention for MI455X (gfx1250, wave32, WMMA).
// S=3, N=20000, E=2000, D=64.  Memory-bound on streaming adj (480MB) ~4x.
// Heavy GEMMs run on v_wmma_f32_16x16x32_bf16 with fp32 accumulation;
// exp(adj) is converted to bf16 on the fly while staging through LDS.
// Staging is fully branch-free: M-boundaries handled by address clamping
// (garbage rows land in discarded C rows), K-boundaries by zero-padding B,
// so every block runs the batched wide-load path; next K tile is prefetched
// with global_prefetch_b8.
// ---------------------------------------------------------------------------

typedef __attribute__((ext_vector_type(16))) __bf16 v16bf;
typedef __attribute__((ext_vector_type(8)))  float  v8f;

#define GNN_N 20000
#define GNN_E 2000
#define GNN_S 3
#define GNN_D 64

__device__ __forceinline__ unsigned short f2bf(float f) {
    unsigned int u = __float_as_uint(f);
    unsigned int r = (u + 0x7FFFu + ((u >> 16) & 1u)) >> 16;   // RNE
    return (unsigned short)r;
}
__device__ __forceinline__ int imin(int a, int b) { return a < b ? a : b; }

// ---------------------------------------------------------------------------
// Fused softmax statistics: row_sum[n] = sum_e exp(adj[n,e]),
//                           col_sum[e] = sum_n exp(adj[n,e]).
// One pass over adj[s].  Each thread owns 4 columns (one float4 per row);
// rows unrolled x5 so loads batch; exp is unconditional (mask multiply),
// wave shfl-reduce for rows.  grid = (2, 250 chunks of 80), block = 256.
// ---------------------------------------------------------------------------
__global__ void stats_kernel(const float* __restrict__ adj,
                             float* __restrict__ row_sum,
                             float* __restrict__ col_sum,
                             int N, int E, int rowChunk) {
    const int t   = threadIdx.x;
    const int nC4 = E >> 2;                         // 500 float4 columns
    const int c4  = blockIdx.x * 256 + t;
    const int cc4 = imin(c4, nC4 - 1);
    const float msk = (c4 < nC4) ? 1.f : 0.f;
    const float4* __restrict__ src = (const float4*)adj;
    const int r0 = blockIdx.y * rowChunk;
    const int r1 = imin(N, r0 + rowChunk);          // rowChunk divides N exactly
    float ca0 = 0.f, ca1 = 0.f, ca2 = 0.f, ca3 = 0.f;
    for (int rb = r0; rb < r1; rb += 5) {
        float4 v[5];
        #pragma unroll
        for (int j = 0; j < 5; ++j) v[j] = src[(rb + j) * nC4 + cc4];
        #pragma unroll
        for (int j = 0; j < 5; ++j) {
            float e0 = __expf(v[j].x) * msk;
            float e1 = __expf(v[j].y) * msk;
            float e2 = __expf(v[j].z) * msk;
            float e3 = __expf(v[j].w) * msk;
            ca0 += e0; ca1 += e1; ca2 += e2; ca3 += e3;
            float s = (e0 + e1) + (e2 + e3);
            #pragma unroll
            for (int off = 16; off > 0; off >>= 1) s += __shfl_down(s, off, 32);
            if ((t & 31) == 0) unsafeAtomicAdd(&row_sum[rb + j], s);
        }
    }
    if (c4 < nC4) {
        unsafeAtomicAdd(&col_sum[4 * c4 + 0], ca0);
        unsafeAtomicAdd(&col_sum[4 * c4 + 1], ca1);
        unsafeAtomicAdd(&col_sum[4 * c4 + 2], ca2);
        unsafeAtomicAdd(&col_sum[4 * c4 + 3], ca3);
    }
}

// ---------------------------------------------------------------------------
// WMMA GEMM:  Cpre[M,64] += sum_k expA[m,k] * B[k,64]   (fp32 atomic K-split)
//   TRANSA=true :  expA[m,k] = exp(adj[k*ld + m])   (adjT_sm products, M=E)
//                  K (= N) is a multiple of 32; k always valid.
//   TRANSA=false:  expA[m,k] = exp(adj[m*ld + k])   (adj_sm product,  M=N)
//                  K is padded to a multiple of 32; B rows >= kA are ZERO,
//                  adj k-address is clamped to kA-4 (zero B kills garbage).
// M-boundaries: load addresses clamped; garbage rows land in C rows >= M,
// which the guarded atomic epilogue discards.  => staging is branch-free.
// block = 256 threads = 8 wave32.  C tile 64x64, K tile 32 (bf16 16x16x32).
// wave w: row sub-tile (w>>1)*16, col group (w&1)*32 -> two 16x16 f32 accs.
// ---------------------------------------------------------------------------
template <bool TRANSA>
__global__ void wmma_gemm_kernel(const float* __restrict__ adj,
                                 const unsigned short* __restrict__ Bb, // bf16 [K,64]
                                 float* __restrict__ Cpre,              // [M,64]
                                 int M, int K, int ld, int kA, int kChunk) {
    __shared__ __align__(16) unsigned short As[64 * 34];   // [m][k] bf16, padded
    __shared__ __align__(16) unsigned short Bt[64 * 34];   // [col][k] bf16, padded

    const int t    = threadIdx.x;
    const int w    = t >> 5;
    const int lane = t & 31;
    const int l    = lane & 15;
    const int hi   = lane >> 4;
    const int m0   = blockIdx.x * 64;
    const int kStart = blockIdx.y * kChunk;
    const int kEnd   = imin(K, kStart + kChunk);    // always a multiple of 32
    const int rTile  = w >> 1;           // 0..3
    const int cGrp   = (w & 1) * 32;     // 0 or 32
    const int M1 = M - 1;
    const int M4 = M - 4;
    const int kA4 = kA - 4;

    v8f acc0 = {};
    v8f acc1 = {};

    for (int kk = kStart; kk < kEnd; kk += 32) {
        // ---------------- stage A tile (exp -> bf16), branch-free ----------------
        if (TRANSA) {
            #pragma unroll
            for (int i = 0; i < 2; ++i) {
                int e4 = t + 256 * i;            // 0..511
                int kr = e4 >> 4;                // 0..31
                int mq = (e4 & 15) << 2;         // 0,4,...,60
                int mAdr = imin(m0 + mq, M4);    // clamp: garbage -> discarded rows
                const float* p = adj + (kk + kr) * ld + mAdr;
                float4 v = *(const float4*)p;
                if (kk + 32 < kEnd) __builtin_prefetch(p + 32 * ld, 0, 1);
                As[(mq + 0) * 34 + kr] = f2bf(__expf(v.x));
                As[(mq + 1) * 34 + kr] = f2bf(__expf(v.y));
                As[(mq + 2) * 34 + kr] = f2bf(__expf(v.z));
                As[(mq + 3) * 34 + kr] = f2bf(__expf(v.w));
            }
        } else {
            #pragma unroll
            for (int i = 0; i < 2; ++i) {
                int e4 = t + 256 * i;
                int mr = e4 >> 3;                // 0..63
                int kq = (e4 & 7) << 2;          // 0,4,...,28
                int mAdr = imin(m0 + mr, M1);    // clamp rows -> discarded C rows
                int kAdr = imin(kk + kq, kA4);   // clamp k -> multiplied by zero B
                const float* p = adj + mAdr * ld + kAdr;
                float4 v = *(const float4*)p;
                if (kk + 32 < kEnd) __builtin_prefetch(p + 32, 0, 1);
                unsigned int p0 = (unsigned int)f2bf(__expf(v.x)) |
                                  ((unsigned int)f2bf(__expf(v.y)) << 16);
                unsigned int p1 = (unsigned int)f2bf(__expf(v.z)) |
                                  ((unsigned int)f2bf(__expf(v.w)) << 16);
                *(unsigned int*)(&As[mr * 34 + kq])     = p0;
                *(unsigned int*)(&As[mr * 34 + kq + 2]) = p1;
            }
        }
        // ---------------- stage B tile transposed (always in-bounds) ------------
        #pragma unroll
        for (int i = 0; i < 4; ++i) {
            int e2 = t + 256 * i;                // 0..1023
            int kr = e2 >> 5;                    // 0..31
            int c2 = (e2 & 31) << 1;             // 0,2,...,62
            unsigned int pv = *(const unsigned int*)(Bb + (kk + kr) * 64 + c2);
            Bt[c2 * 34 + kr]       = (unsigned short)pv;
            Bt[(c2 + 1) * 34 + kr] = (unsigned short)(pv >> 16);
        }
        __syncthreads();

        // ---- build fragments per documented 16-bit WMMA layouts ----
        union Frag { v16bf v; unsigned int u[8]; } aF, bF0, bF1;
        const unsigned int* Arow  = (const unsigned int*)(As + (rTile * 16 + l) * 34);
        const unsigned int* Brow0 = (const unsigned int*)(Bt + (cGrp + l) * 34);
        const unsigned int* Brow1 = (const unsigned int*)(Bt + (cGrp + 16 + l) * 34);
        #pragma unroll
        for (int g = 0; g < 8; ++g) {
            // A 16x32: VGPR g<4 -> K = 2g + 8*hi ; g>=4 -> K = 16 + 2(g-4) + 8*hi
            int kpA = (g < 4) ? (2 * g + 8 * hi) : (16 + 2 * (g - 4) + 8 * hi);
            aF.u[g] = Arow[kpA >> 1];
            // B 32x16: same K order as A, lane = column
            int kpB = kpA;
            bF0.u[g] = Brow0[kpB >> 1];
            bF1.u[g] = Brow1[kpB >> 1];
        }
        acc0 = __builtin_amdgcn_wmma_f32_16x16x32_bf16(false, aF.v, false, bF0.v,
                                                       (short)0, acc0, false, false);
        acc1 = __builtin_amdgcn_wmma_f32_16x16x32_bf16(false, aF.v, false, bF1.v,
                                                       (short)0, acc1, false, false);
        __syncthreads();
    }

    // ---- K-split accumulation into global fp32 (discards clamped rows) ----
    #pragma unroll
    for (int r = 0; r < 8; ++r) {
        int gm = m0 + rTile * 16 + r + 8 * hi;   // C layout: VGPR r -> M = r + 8*hi
        if (gm < M) {
            unsafeAtomicAdd(&Cpre[gm * 64 + cGrp + l],      acc0[r]);
            unsafeAtomicAdd(&Cpre[gm * 64 + cGrp + 16 + l], acc1[r]);
        }
    }
}

// ---------------------------------------------------------------------------
// x = fe @ W2 -> bf16.  block 256 handles 4 rows; W2 cached in LDS.
// ---------------------------------------------------------------------------
__global__ void proj_kernel(const float* __restrict__ fe, const float* __restrict__ W,
                            unsigned short* __restrict__ xb, int N) {
    __shared__ float Ws[64 * 64];
    __shared__ float Fs[256];
    int t = threadIdx.x;
    for (int i = t; i < 4096; i += 256) Ws[i] = W[i];
    int nl = t >> 6, d = t & 63;
    int n = blockIdx.x * 4 + nl;
    int nc = imin(n, N - 1);
    Fs[t] = fe[nc * 64 + d];
    __syncthreads();
    if (n < N) {
        float acc = 0.f;
        #pragma unroll
        for (int k = 0; k < 64; ++k) acc += Fs[nl * 64 + k] * Ws[k * 64 + d];
        xb[n * 64 + d] = f2bf(acc);
    }
}

// ---------------------------------------------------------------------------
// edge = relu(edge_pre / col_sum);  y = edge @ W3 -> bf16
// ---------------------------------------------------------------------------
__global__ void edge_y_kernel(const float* __restrict__ edge_pre,
                              const float* __restrict__ col_sum,
                              const float* __restrict__ W3,
                              unsigned short* __restrict__ yb, int E) {
    __shared__ float Ws[64 * 64];
    __shared__ float Es[256];
    int t = threadIdx.x;
    for (int i = t; i < 4096; i += 256) Ws[i] = W3[i];
    int el = t >> 6, d = t & 63;
    int e = blockIdx.x * 4 + el;
    int ec = imin(e, E - 1);
    Es[t] = fmaxf(edge_pre[ec * 64 + d] / col_sum[ec], 0.f);
    __syncthreads();
    if (e < E) {
        float acc = 0.f;
        #pragma unroll
        for (int k = 0; k < 64; ++k) acc += Es[el * 64 + k] * Ws[k * 64 + d];
        yb[e * 64 + d] = f2bf(acc);
    }
}

// node = relu(node_pre / row_sum) -> f32 + bf16
__global__ void node_epi_kernel(const float* __restrict__ node_pre,
                                const float* __restrict__ row_sum,
                                float* __restrict__ node,
                                unsigned short* __restrict__ nodeb, int N) {
    int idx = blockIdx.x * 256 + threadIdx.x;
    if (idx >= N * 64) return;
    int n = idx >> 6;
    float v = fmaxf(node_pre[idx] / row_sum[n], 0.f);
    node[idx] = v;
    nodeb[idx] = f2bf(v);
}

// out_edges = BN(relu(eo_pre / col_sum))
__global__ void eo_epi_kernel(const float* __restrict__ eo_pre,
                              const float* __restrict__ col_sum,
                              const float* __restrict__ gamma,
                              const float* __restrict__ beta,
                              float* __restrict__ out_edges, int E) {
    int idx = blockIdx.x * 256 + threadIdx.x;
    if (idx >= E * 64) return;
    int e = idx >> 6, d = idx & 63;
    float v = fmaxf(eo_pre[idx] / col_sum[e], 0.f);
    v = v * (gamma[d] * rsqrtf(1.f + 1e-5f)) + beta[d];
    out_edges[idx] = v;
}

// ---------------------------------------------------------------------------
// Fusion: emb = [fe, BN(node)]; z_i = tanh(emb_i@w1+b1)@w2+b2; p = softmax_i;
// fe' = p0*fe + p1*BN(node).  Writes fe in place + out nodes slice.
// block 256 handles 16 nodes; w1 cached in LDS.
// ---------------------------------------------------------------------------
__global__ void fusion_kernel(float* __restrict__ fe,
                              const float* __restrict__ node,
                              const float* __restrict__ gamma,
                              const float* __restrict__ beta,
                              const float* __restrict__ w1,
                              const float* __restrict__ b1,
                              const float* __restrict__ w2,
                              const float* __restrict__ b2,
                              float* __restrict__ out_nodes, int N) {
    __shared__ float W1s[64 * 64];
    __shared__ float w2s[64], b1s[64];
    __shared__ float F[16 * 64], G[16 * 64];
    __shared__ float Sx[256];
    __shared__ float sc[16][2];
    int t = threadIdx.x;
    for (int i = t; i < 4096; i += 256) W1s[i] = w1[i];
    if (t < 64) { w2s[t] = w2[t]; b1s[t] = b1[t]; }
    int n0 = blockIdx.x * 16;
    float bnmul = rsqrtf(1.f + 1e-5f);
    for (int i = 0; i < 4; ++i) {
        int idx = t + 256 * i;                 // 0..1023
        int nl = idx >> 6, d = idx & 63;
        int n = n0 + nl;
        int nc = imin(n, N - 1);
        F[idx] = fe[nc * 64 + d];
        G[idx] = node[nc * 64 + d] * (gamma[d] * bnmul) + beta[d];
    }
    __syncthreads();
    float b2v = b2[0];
    for (int i = 0; i < 4; ++i) {
        int c = t & 63, var = (t >> 6) & 1, nl = (t >> 7) + 2 * i;
        const float* row = (var ? G : F) + nl * 64;
        float acc = b1s[c];
        #pragma unroll
        for (int d = 0; d < 64; ++d) acc += row[d] * W1s[d * 64 + c];
        Sx[t] = tanhf(acc) * w2s[c];
        __syncthreads();
        if (t < 4) {
            float s = 0.f;
            #pragma unroll
            for (int j = 0; j < 64; ++j) s += Sx[t * 64 + j];
            sc[(t >> 1) + 2 * i][t & 1] = s + b2v;
        }
        __syncthreads();
    }
    for (int i = 0; i < 4; ++i) {
        int idx = t + 256 * i;
        int nl = idx >> 6, d = idx & 63;
        int n = n0 + nl;
        if (n < N) {
            float z0 = sc[nl][0], z1 = sc[nl][1];
            float m  = fmaxf(z0, z1);
            float e0 = __expf(z0 - m), e1 = __expf(z1 - m);
            float p0 = e0 / (e0 + e1);
            float v  = p0 * F[idx] + (1.f - p0) * G[idx];
            fe[n * 64 + d] = v;
            out_nodes[n * 64 + d] = v;
        }
    }
}

// ---------------------------------------------------------------------------
extern "C" void kernel_launch(void* const* d_in, const int* in_sizes, int n_in,
                              void* d_out, int out_size, void* d_ws, size_t ws_size,
                              hipStream_t stream) {
    const int N = GNN_N, E = GNN_E, S = GNN_S;
    const int Kpad = 2048;                 // E padded up to a multiple of 32
    const float* fe0   = (const float*)d_in[0];
    const float* adj   = (const float*)d_in[1];
    const float* W2    = (const float*)d_in[2];
    const float* W3    = (const float*)d_in[3];
    const float* gamma = (const float*)d_in[4];
    const float* beta  = (const float*)d_in[5];
    const float* fw1   = (const float*)d_in[6];
    const float* fb1   = (const float*)d_in[7];
    const float* fw2   = (const float*)d_in[8];
    const float* fb2   = (const float*)d_in[9];
    float* out = (float*)d_out;

    // carve workspace (zeroed region first: accumulators, sums, padded yb)
    char* ws = (char*)d_ws;
    size_t off = 0;
    auto carve = [&](size_t bytes) -> void* {
        void* p = ws + off;
        off = (off + bytes + 255) & ~(size_t)255;
        return p;
    };
    float* edge_pre = (float*)carve((size_t)E * 64 * 4);
    float* eo_pre   = (float*)carve((size_t)E * 64 * 4);
    float* node_pre = (float*)carve((size_t)N * 64 * 4);
    float* row_sum  = (float*)carve((size_t)N * 4);
    float* col_sum  = (float*)carve((size_t)E * 4);
    unsigned short* yb = (unsigned short*)carve((size_t)Kpad * 64 * 2); // rows >=E stay 0
    size_t zeroBytes = off;
    float*          fe_cur = (float*)carve((size_t)N * 64 * 4);
    float*          node   = (float*)carve((size_t)N * 64 * 4);
    unsigned short* xb     = (unsigned short*)carve((size_t)N * 64 * 2);
    unsigned short* nodeb  = (unsigned short*)carve((size_t)N * 64 * 2);

    hipMemcpyAsync(fe_cur, fe0, (size_t)N * 64 * 4, hipMemcpyDeviceToDevice, stream);

    for (int s = 0; s < S; ++s) {
        const float* adjS = adj + (size_t)s * N * E;
        float* outN = out + (size_t)s * (N + E) * 64;
        float* outE = outN + (size_t)N * 64;

        hipMemsetAsync(ws, 0, zeroBytes, stream);

        // softmax denominators (one pass over adj[s])
        stats_kernel<<<dim3(2, 250), 256, 0, stream>>>(adjS, row_sum, col_sum, N, E, 80);

        // x = fe @ W2 -> bf16
        proj_kernel<<<(N + 3) / 4, 256, 0, stream>>>(fe_cur, W2, xb, N);

        // edge_pre += exp(adj)^T @ x   (M=E, K=N multiple of 32, 16-way K-split)
        wmma_gemm_kernel<true><<<dim3((E + 63) / 64, 16), 256, 0, stream>>>(
            adjS, xb, edge_pre, E, N, E, N, 1280);

        // edge = relu(edge_pre/col_sum); y = edge @ W3 -> bf16 (rows >=E are zero)
        edge_y_kernel<<<(E + 3) / 4, 256, 0, stream>>>(edge_pre, col_sum, W3, yb, E);

        // node_pre += exp(adj) @ y   (M=N, K padded to 2048, 4-way K-split)
        wmma_gemm_kernel<false><<<dim3((N + 63) / 64, 4), 256, 0, stream>>>(
            adjS, yb, node_pre, N, Kpad, E, E, 512);

        // node = relu(node_pre/row_sum) -> f32 + bf16
        node_epi_kernel<<<(N * 64 + 255) / 256, 256, 0, stream>>>(node_pre, row_sum,
                                                                  node, nodeb, N);

        // eo_pre += exp(adj)^T @ node
        wmma_gemm_kernel<true><<<dim3((E + 63) / 64, 16), 256, 0, stream>>>(
            adjS, nodeb, eo_pre, E, N, E, N, 1280);

        // out edges slice = BN(relu(eo_pre/col_sum))
        eo_epi_kernel<<<(E * 64 + 255) / 256, 256, 0, stream>>>(eo_pre, col_sum,
                                                                gamma, beta, outE, E);

        // fusion: fe = softmax-gated mix of fe and BN(node); also writes out nodes
        fusion_kernel<<<(N + 15) / 16, 256, 0, stream>>>(fe_cur, node, gamma, beta,
                                                         fw1, fb1, fw2, fb2, outN, N);
    }
    (void)in_sizes; (void)n_in; (void)out_size; (void)ws_size;
}
